// TextDecoderTensorCache_16973710754309
// MI455X (gfx1250) — matrix operand order repaired and checked by reading, not compiled, not run on hardware
//
#include <hip/hip_runtime.h>
#include <hip/hip_bf16.h>

// ---------------------------------------------------------------------------
// Whisper text-decoder single-token step, batch=8, d=768, 12 layers.
// Memory-bound (~1.5 GB/call -> ~65us floor at 23.3 TB/s). All GEMMs are M=8
// GEMVs padded to M=16 WMMA tiles (v_wmma_f32_16x16x32_f16). Weights stream
// once as global_load_b128, software-pipelined through double-buffered LDS;
// per-lane fragments are contiguous ds_load_b128.
// ---------------------------------------------------------------------------

#define D_MODEL   768
#define N_HEAD    12
#define D_HEAD    64
#define N_LAYER   12
#define BATCH     8
#define N_CTX     448
#define N_AUDIO   1500
#define D_FF      3072
#define N_VOCAB   51865

typedef __attribute__((ext_vector_type(16))) _Float16 v16h;
typedef __attribute__((ext_vector_type(8)))  _Float16 v8h;
typedef __attribute__((ext_vector_type(4)))  _Float16 v4h;
typedef __attribute__((ext_vector_type(2)))  _Float16 v2h;
typedef __attribute__((ext_vector_type(8)))  float    v8f;

#define GEMM_WAVES 4
#define GEMM_COLS  (GEMM_WAVES * 16)   // 64 output columns per block
#define SA_STRIDE  40                  // 80B rows: 16B aligned, bank-spread
#define SB_STRIDE  40
#define ACT_NONE 0
#define ACT_GELU 1

__device__ __forceinline__ float f4c(const float4& v, int j) {
  return j == 0 ? v.x : j == 1 ? v.y : j == 2 ? v.z : v.w;
}
__device__ __forceinline__ v4h cvt4(const float4& v) {
  v4h r;
  r[0] = (_Float16)v.x; r[1] = (_Float16)v.y;
  r[2] = (_Float16)v.z; r[3] = (_Float16)v.w;
  return r;
}

// ---------------------------------------------------------------------------
// out[8,N] = act(A[8,K] @ W + bias) + res
// BT=false: W is [K,N] (torch Linear [in,out]).  BT=true: W is [N,K] (embed).
// One wave owns one 16-col WMMA tile; A padded M 8->16 with zero rows.
// Software-pipelined: store chunk i to LDS, issue chunk i+1 b128 loads,
// barrier, WMMA chunk i. One barrier per 32-wide K chunk (double buffer).
// ---------------------------------------------------------------------------
template <bool BT, int K, int N>
__global__ void __launch_bounds__(GEMM_WAVES * 32)
gemm8_wmma(const float* __restrict__ A,
           const float* __restrict__ W,
           const float* __restrict__ bias,
           const float* __restrict__ res,
           float* __restrict__ out, int act) {
  __shared__ __align__(16) _Float16 sA[2][16 * SA_STRIDE];        // [row][k]
  __shared__ __align__(16) _Float16 sB[2][GEMM_COLS * SB_STRIDE]; // [n][k]

  const int tid  = threadIdx.x;
  const int lane = tid & 31;
  const int wave = tid >> 5;
  const int n0   = blockIdx.x * GEMM_COLS;

  // zero only the M-pad rows 8..15 (disjoint from staged rows 0..7; becomes
  // visible to all threads at the first in-loop barrier)
  for (int i = 8 * SA_STRIDE + tid; i < 16 * SA_STRIDE; i += GEMM_WAVES * 32) {
    sA[0][i] = (_Float16)0.f;
    sA[1][i] = (_Float16)0.f;
  }

  // staging coordinates
  const int a_r  = tid >> 3, a_kf = tid & 7;   // A: 8 rows x 8 float4 (tid<64)
  const int b_nf = tid & 15, b_kp = tid >> 4;  // KN: 16 nf x 8 k-pairs
  const int b_nn = tid >> 3, b_kf = tid & 7;   // BT: 16 n x 8 float4-of-k

  const float4* A4 = (const float4*)A;
  const float4* W4 = (const float4*)W;

  float4 ra, rb0, rb1, rb2, rb3;

  auto load_chunk = [&](int k0) {
    if (tid < 64) ra = A4[a_r * (K / 4) + (k0 >> 2) + a_kf];
    if (!BT) {
      const int base = (n0 >> 2) + b_nf;
      rb0 = W4[(k0 + 2 * b_kp)      * (N / 4) + base];
      rb1 = W4[(k0 + 2 * b_kp + 1)  * (N / 4) + base];
      rb2 = W4[(k0 + 2 * b_kp + 16) * (N / 4) + base];
      rb3 = W4[(k0 + 2 * b_kp + 17) * (N / 4) + base];
    } else {
      const float4 z = make_float4(0.f, 0.f, 0.f, 0.f);
      const int kb = (k0 >> 2) + b_kf;
      rb0 = (n0 + b_nn      < N) ? W4[(n0 + b_nn)      * (K / 4) + kb] : z;
      rb1 = (n0 + b_nn + 16 < N) ? W4[(n0 + b_nn + 16) * (K / 4) + kb] : z;
      rb2 = (n0 + b_nn + 32 < N) ? W4[(n0 + b_nn + 32) * (K / 4) + kb] : z;
      rb3 = (n0 + b_nn + 48 < N) ? W4[(n0 + b_nn + 48) * (K / 4) + kb] : z;
    }
  };

  auto store_chunk = [&](int p) {
    if (tid < 64)
      *(v4h*)&sA[p][a_r * SA_STRIDE + 4 * a_kf] = cvt4(ra);
    if (!BT) {
      // transpose: pack (k, k+1) pairs as half2 -> ds_store_b32 per n
#pragma unroll
      for (int j = 0; j < 4; ++j) {
        const int n = b_nf * 4 + j;
        v2h lo; lo[0] = (_Float16)f4c(rb0, j); lo[1] = (_Float16)f4c(rb1, j);
        v2h hi; hi[0] = (_Float16)f4c(rb2, j); hi[1] = (_Float16)f4c(rb3, j);
        *(v2h*)&sB[p][n * SB_STRIDE + 2 * b_kp]      = lo;
        *(v2h*)&sB[p][n * SB_STRIDE + 2 * b_kp + 16] = hi;
      }
    } else {
      // k-contiguous in memory: straight b64 stores
      *(v4h*)&sB[p][(b_nn)      * SB_STRIDE + 4 * b_kf] = cvt4(rb0);
      *(v4h*)&sB[p][(b_nn + 16) * SB_STRIDE + 4 * b_kf] = cvt4(rb1);
      *(v4h*)&sB[p][(b_nn + 32) * SB_STRIDE + 4 * b_kf] = cvt4(rb2);
      *(v4h*)&sB[p][(b_nn + 48) * SB_STRIDE + 4 * b_kf] = cvt4(rb3);
    }
  };

  // per-lane fragment coordinates (ISA 16-bit WMMA layouts, wave32)
  const int ar  = lane & 15;
  const int akb = (lane >> 4) * 8;
  const int bn  = wave * 16 + (lane & 15);
  const int bkb = (lane >> 4) * 8;

  v8f c = {};
  load_chunk(0);
  int p = 0;
#pragma unroll 2
  for (int k0 = 0; k0 < K; k0 += 32) {
    store_chunk(p);
    if (k0 + 32 < K) load_chunk(k0 + 32);   // overlaps barrier + WMMA below
    __syncthreads();

    const _Float16* ap = &sA[p][ar * SA_STRIDE + akb];
    const v8h alo = *(const v8h*)ap;
    const v8h ahi = *(const v8h*)(ap + 16);
    const v16h af = __builtin_shufflevector(alo, ahi, 0, 1, 2, 3, 4, 5, 6, 7,
                                            8, 9, 10, 11, 12, 13, 14, 15);
    const _Float16* bp = &sB[p][bn * SB_STRIDE + bkb];
    const v8h blo = *(const v8h*)bp;
    const v8h bhi = *(const v8h*)(bp + 16);
    const v16h bf = __builtin_shufflevector(blo, bhi, 0, 1, 2, 3, 4, 5, 6, 7,
                                            8, 9, 10, 11, 12, 13, 14, 15);

    c = __builtin_amdgcn_wmma_f32_16x16x32_f16(false, af, false, bf,
                                               (short)0, c, false, false);
    p ^= 1;   // single barrier per chunk is safe with double buffering
  }

  // ---- epilogue: D layout -> VGPR r, lanes 0-15 hold M=r (rows 8..15 pad)
  if (lane < 16) {
    const int n = n0 + wave * 16 + lane;
    if (n < N) {
      const float bv = bias ? bias[n] : 0.f;
#pragma unroll
      for (int r = 0; r < 8; ++r) {
        float val = c[r] + bv;
        if (act == ACT_GELU)
          val = 0.5f * val * (1.f + erff(val * 0.70710678118654752f));
        if (res) val += res[r * N + n];
        out[r * N + n] = val;
      }
    }
  }
}

// ---------------------------------------------------------------------------
// LayerNorm over 768, one block per row (8 rows)
// ---------------------------------------------------------------------------
__global__ void ln_kernel(const float* __restrict__ x,
                          const float* __restrict__ g,
                          const float* __restrict__ b,
                          float* __restrict__ out) {
  const int row = blockIdx.x, tid = threadIdx.x;
  __shared__ float red[256];
  const float* xr = x + row * D_MODEL;

  float sm = 0.f;
  for (int d = tid; d < D_MODEL; d += 256) sm += xr[d];
  red[tid] = sm; __syncthreads();
  for (int o = 128; o > 0; o >>= 1) { if (tid < o) red[tid] += red[tid + o]; __syncthreads(); }
  const float mean = red[0] * (1.f / D_MODEL);
  __syncthreads();

  float vs = 0.f;
  for (int d = tid; d < D_MODEL; d += 256) { float t = xr[d] - mean; vs += t * t; }
  red[tid] = vs; __syncthreads();
  for (int o = 128; o > 0; o >>= 1) { if (tid < o) red[tid] += red[tid + o]; __syncthreads(); }
  const float rstd = rsqrtf(red[0] * (1.f / D_MODEL) + 1e-5f);

  for (int d = tid; d < D_MODEL; d += 256)
    out[row * D_MODEL + d] = (xr[d] - mean) * rstd * g[d] + b[d];
}

// ---------------------------------------------------------------------------
// Attention for tq=1: one block per (head, batch). Two-pass softmax in LDS.
// Self-attn: Kc/Vc are the layer's input cache with shift=1, plus knew/vnew
// as the virtual last slot (matches concat(cache[:,1:], new)).
// K/V rows are streamed as float4 (global_load_b128).
// ---------------------------------------------------------------------------
__global__ void attn_kernel(const float* __restrict__ q,
                            const float* __restrict__ Kc,
                            const float* __restrict__ Vc,
                            const float* __restrict__ knew,
                            const float* __restrict__ vnew,
                            const float* __restrict__ mask,
                            float* __restrict__ out,
                            int T, int Tstride, int shift) {
  const int h = blockIdx.x, b = blockIdx.y, tid = threadIdx.x;
  const int d0 = h * D_HEAD;
  __shared__ float qs[D_HEAD];
  __shared__ float s[N_AUDIO];
  __shared__ float red[256];
  __shared__ __align__(16) float pv[16][D_HEAD];

  // q scaled by dh^-0.25 on both q and k  ->  qk * dh^-0.5 = 0.125
  if (tid < D_HEAD) qs[tid] = q[b * D_MODEL + d0 + tid] * 0.125f;
  __syncthreads();

  // ---- scores
  for (int t = tid; t < T; t += 256) {
    const float* krow = (knew && t == T - 1)
        ? knew + b * D_MODEL + d0
        : Kc + ((size_t)b * Tstride + (t + shift)) * D_MODEL + d0;
    const float4* k4 = (const float4*)krow;   // 256B-aligned rows
    float acc = 0.f;
#pragma unroll
    for (int d4 = 0; d4 < D_HEAD / 4; ++d4) {
      const float4 kv = k4[d4];
      acc += qs[4 * d4 + 0] * kv.x + qs[4 * d4 + 1] * kv.y +
             qs[4 * d4 + 2] * kv.z + qs[4 * d4 + 3] * kv.w;
    }
    s[t] = acc + (mask ? mask[t] : 0.f);
  }
  __syncthreads();

  // ---- softmax (two-pass)
  float lm = -3.4e38f;
  for (int t = tid; t < T; t += 256) lm = fmaxf(lm, s[t]);
  red[tid] = lm; __syncthreads();
  for (int o = 128; o > 0; o >>= 1) { if (tid < o) red[tid] = fmaxf(red[tid], red[tid + o]); __syncthreads(); }
  const float m = red[0];
  __syncthreads();

  float ls = 0.f;
  for (int t = tid; t < T; t += 256) { float e = __expf(s[t] - m); s[t] = e; ls += e; }
  red[tid] = ls; __syncthreads();
  for (int o = 128; o > 0; o >>= 1) { if (tid < o) red[tid] += red[tid + o]; __syncthreads(); }
  const float inv = 1.f / red[0];

  // ---- weighted V: 16 t-chunks x 16 lanes x float4, coalesced b128 streams
  const int tc = tid >> 4;          // 0..15 time chunk
  const int dq = tid & 15;          // float4 index within the 64-dim head
  float4 a4 = make_float4(0.f, 0.f, 0.f, 0.f);
  for (int t = tc; t < T; t += 16) {
    const float* vrow = (vnew && t == T - 1)
        ? vnew + b * D_MODEL + d0
        : Vc + ((size_t)b * Tstride + (t + shift)) * D_MODEL + d0;
    const float4 vv = ((const float4*)vrow)[dq];
    const float w = s[t];
    a4.x += w * vv.x; a4.y += w * vv.y; a4.z += w * vv.z; a4.w += w * vv.w;
  }
  ((float4*)pv[tc])[dq] = a4;
  __syncthreads();
  if (tid < D_HEAD) {
    float o = 0.f;
#pragma unroll
    for (int cc = 0; cc < 16; ++cc) o += pv[cc][tid];
    out[b * D_MODEL + d0 + tid] = o * inv;
  }
}

// ---------------------------------------------------------------------------
// Embedding gather + positional add
// ---------------------------------------------------------------------------
__global__ void embed_kernel(const int* __restrict__ tokens,
                             const float* __restrict__ emb,
                             const float* __restrict__ pos,
                             float* __restrict__ x) {
  const int b = blockIdx.x, tid = threadIdx.x;
  const int tok = tokens[b];
  for (int d = tid; d < D_MODEL; d += 256)
    x[b * D_MODEL + d] = emb[(size_t)tok * D_MODEL + d] + pos[d];
}

// ---------------------------------------------------------------------------
// Roll both KV caches for all layers: out[l,b,t,:] = in[l,b,t+1,:], t<447.
// float4 grid-stride; ~528 MB of the traffic lives here.
// ---------------------------------------------------------------------------
__global__ void cache_shift_kernel(const float4* __restrict__ kin,
                                   const float4* __restrict__ vin,
                                   float4* __restrict__ kout,
                                   float4* __restrict__ vout) {
  const size_t ROWV = D_MODEL / 4;  // 192
  const size_t total = (size_t)N_LAYER * BATCH * (N_CTX - 1) * ROWV;
  for (size_t i = (size_t)blockIdx.x * blockDim.x + threadIdx.x; i < total;
       i += (size_t)gridDim.x * blockDim.x) {
    size_t d   = i % ROWV;
    size_t row = i / ROWV;
    size_t t   = row % (N_CTX - 1);
    size_t lb  = row / (N_CTX - 1);
    size_t src = (lb * N_CTX + t + 1) * ROWV + d;
    size_t dst = (lb * N_CTX + t) * ROWV + d;
    kout[dst] = kin[src];
    vout[dst] = vin[src];
  }
}

// new k/v into slot 447 of this layer's output cache
__global__ void kv_store_kernel(const float* __restrict__ knew,
                                const float* __restrict__ vnew,
                                float* __restrict__ koutL,
                                float* __restrict__ voutL) {
  const int i = blockIdx.x * blockDim.x + threadIdx.x;
  if (i < BATCH * D_MODEL) {
    const int b = i / D_MODEL, d = i % D_MODEL;
    const size_t off = ((size_t)b * N_CTX + (N_CTX - 1)) * D_MODEL + d;
    koutL[off] = knew[i];
    voutL[off] = vnew[i];
  }
}

// ---------------------------------------------------------------------------
extern "C" void kernel_launch(void* const* d_in, const int* in_sizes, int n_in,
                              void* d_out, int out_size, void* d_ws, size_t ws_size,
                              hipStream_t stream) {
  // inputs in setup_inputs() dict order (params dict flattened in order)
  const int*   tokens = (const int*)d_in[0];
  const float* selfK  = (const float*)d_in[1];
  const float* selfV  = (const float*)d_in[2];
  const float* crossK = (const float*)d_in[3];
  const float* crossV = (const float*)d_in[4];
  const float* pos    = (const float*)d_in[5];
  const float* mask   = (const float*)d_in[6];
  const float* Wq   = (const float*)d_in[7];
  const float* bq   = (const float*)d_in[8];
  const float* Wk   = (const float*)d_in[9];
  const float* Wv   = (const float*)d_in[10];
  const float* bv   = (const float*)d_in[11];
  const float* Wo   = (const float*)d_in[12];
  const float* bo   = (const float*)d_in[13];
  const float* cWq  = (const float*)d_in[14];
  const float* cbq  = (const float*)d_in[15];
  const float* cWo  = (const float*)d_in[16];
  const float* cbo  = (const float*)d_in[17];
  const float* ln1g = (const float*)d_in[18];
  const float* ln1b = (const float*)d_in[19];
  const float* ln2g = (const float*)d_in[20];
  const float* ln2b = (const float*)d_in[21];
  const float* ln3g = (const float*)d_in[22];
  const float* ln3b = (const float*)d_in[23];
  const float* W1   = (const float*)d_in[24];
  const float* b1   = (const float*)d_in[25];
  const float* W2   = (const float*)d_in[26];
  const float* b2   = (const float*)d_in[27];
  const float* lnfg = (const float*)d_in[28];
  const float* lnfb = (const float*)d_in[29];
  const float* emb  = (const float*)d_in[30];

  float* outp   = (float*)d_out;
  float* logits = outp;                                     // [8, 51865]
  float* kout   = logits + (size_t)BATCH * N_VOCAB;         // [12,8,448,768]
  float* vout   = kout + (size_t)N_LAYER * BATCH * N_CTX * D_MODEL;

  // workspace layout (fp32)
  float* ws  = (float*)d_ws;
  float* x   = ws;                      // [8,768]
  float* h   = x + BATCH * D_MODEL;     // [8,768]
  float* q   = h + BATCH * D_MODEL;     // [8,768]
  float* k   = q + BATCH * D_MODEL;     // [8,768]
  float* v   = k + BATCH * D_MODEL;     // [8,768]
  float* ao  = v + BATCH * D_MODEL;     // [8,768]
  float* mid = ao + BATCH * D_MODEL;    // [8,3072]

  const size_t sliceSelf  = (size_t)BATCH * N_CTX * D_MODEL;
  const size_t sliceCross = (size_t)BATCH * N_AUDIO * D_MODEL;
  const size_t wD  = (size_t)D_MODEL * D_MODEL;
  const size_t wF1 = (size_t)D_MODEL * D_FF;

  const dim3 gAttn(N_HEAD, BATCH);
  const int gN   = D_MODEL / GEMM_COLS;                     // 12
  const int gFF  = D_FF / GEMM_COLS;                        // 48
  const int gVoc = (N_VOCAB + GEMM_COLS - 1) / GEMM_COLS;   // 811
  const int gemmT = GEMM_WAVES * 32;                        // 128 threads

  // independent 528 MB cache roll -> launch first
  cache_shift_kernel<<<2048, 256, 0, stream>>>(
      (const float4*)selfK, (const float4*)selfV, (float4*)kout, (float4*)vout);

  embed_kernel<<<BATCH, 256, 0, stream>>>(tokens, emb, pos, x);

  for (int l = 0; l < N_LAYER; ++l) {
    float* koutL = kout + (size_t)l * sliceSelf;
    float* voutL = vout + (size_t)l * sliceSelf;

    // ---- self-attention
    ln_kernel<<<BATCH, 256, 0, stream>>>(x, ln1g + l * D_MODEL, ln1b + l * D_MODEL, h);
    gemm8_wmma<false, D_MODEL, D_MODEL><<<gN, gemmT, 0, stream>>>(
        h, Wq + l * wD, bq + l * D_MODEL, nullptr, q, ACT_NONE);
    gemm8_wmma<false, D_MODEL, D_MODEL><<<gN, gemmT, 0, stream>>>(
        h, Wk + l * wD, nullptr, nullptr, k, ACT_NONE);
    gemm8_wmma<false, D_MODEL, D_MODEL><<<gN, gemmT, 0, stream>>>(
        h, Wv + l * wD, bv + l * D_MODEL, nullptr, v, ACT_NONE);
    kv_store_kernel<<<(BATCH * D_MODEL + 255) / 256, 256, 0, stream>>>(k, v, koutL, voutL);
    attn_kernel<<<gAttn, 256, 0, stream>>>(q, selfK + (size_t)l * sliceSelf,
                                           selfV + (size_t)l * sliceSelf,
                                           k, v, mask, ao, N_CTX, N_CTX, 1);
    gemm8_wmma<false, D_MODEL, D_MODEL><<<gN, gemmT, 0, stream>>>(
        ao, Wo + l * wD, bo + l * D_MODEL, x, x, ACT_NONE);

    // ---- cross-attention
    ln_kernel<<<BATCH, 256, 0, stream>>>(x, ln2g + l * D_MODEL, ln2b + l * D_MODEL, h);
    gemm8_wmma<false, D_MODEL, D_MODEL><<<gN, gemmT, 0, stream>>>(
        h, cWq + l * wD, cbq + l * D_MODEL, nullptr, q, ACT_NONE);
    attn_kernel<<<gAttn, 256, 0, stream>>>(q, crossK + (size_t)l * sliceCross,
                                           crossV + (size_t)l * sliceCross,
                                           nullptr, nullptr, nullptr, ao,
                                           N_AUDIO, N_AUDIO, 0);
    gemm8_wmma<false, D_MODEL, D_MODEL><<<gN, gemmT, 0, stream>>>(
        ao, cWo + l * wD, cbo + l * D_MODEL, x, x, ACT_NONE);

    // ---- MLP (exact GELU)
    ln_kernel<<<BATCH, 256, 0, stream>>>(x, ln3g + l * D_MODEL, ln3b + l * D_MODEL, h);
    gemm8_wmma<false, D_MODEL, D_FF><<<gFF, gemmT, 0, stream>>>(
        h, W1 + l * wF1, b1 + l * D_FF, nullptr, mid, ACT_GELU);
    gemm8_wmma<false, D_FF, D_MODEL><<<gN, gemmT, 0, stream>>>(
        mid, W2 + l * wF1, b2 + l * D_MODEL, x, x, ACT_NONE);
  }

  // ---- final LN + tied-embedding logits (embed is [vocab, d] -> BT layout)
  ln_kernel<<<BATCH, 256, 0, stream>>>(x, lnfg, lnfb, h);
  gemm8_wmma<true, D_MODEL, N_VOCAB><<<gVoc, gemmT, 0, stream>>>(
      h, emb, nullptr, nullptr, logits, ACT_NONE);
}
